// HigherOrderAttention_21723944583531
// MI455X (gfx1250) — matrix operand order repaired and checked by reading, not compiled, not run on hardware
//
#include <hip/hip_runtime.h>
#include <stdint.h>

typedef unsigned short u16;
typedef unsigned int   u32;

#define BATCH 2
#define NSEQ  192
#define DIMSZ 512
#define HEADS 8
#define DH    64
#define QKV_COLS 2560
#define ITILES 12      // NSEQ / 16
#define WAVES 4
#define KT    32       // key-pairs per inner step

typedef __attribute__((ext_vector_type(16))) __bf16 v16bf;
typedef __attribute__((ext_vector_type(8)))  float  v8f;

union BfOp { v16bf v; uint4 q[2]; };

// ---------- bf16 helpers ----------
__device__ __forceinline__ float bflo(u32 a) {
    union { u32 i; float f; } c; c.i = a << 16; return c.f;
}
__device__ __forceinline__ u16 f2bf(float f) {   // manual RNE (cold paths only)
    union { float f; u32 i; } c; c.f = f;
    u32 i = c.i;
    u32 r = (i + 0x7fffu + ((i >> 16) & 1u)) >> 16;
    return (u16)r;
}
// packed bf16 multiply: 2 elements / instruction (VOP3P, CDNA5)
__device__ __forceinline__ u32 pkmul_bf16(u32 a, u32 b) {
    u32 d;
    asm("v_pk_mul_bf16 %0, %1, %2" : "=v"(d) : "v"(a), "v"(b));
    return d;
}
// pack two f32 -> 2x bf16 (RNE) in one instruction
__device__ __forceinline__ u32 cvt_pk_bf16(float lo, float hi) {
    u32 d;
    asm("v_cvt_pk_bf16_f32 %0, %1, %2" : "=v"(d) : "v"(lo), "v"(hi));
    return d;
}
__device__ __forceinline__ uint4 pkmul8(uint4 a, uint4 b) {
    return make_uint4(pkmul_bf16(a.x, b.x), pkmul_bf16(a.y, b.y),
                      pkmul_bf16(a.z, b.z), pkmul_bf16(a.w, b.w));
}

// XOR butterfly within 16-lane rows via v_permlane16_b32 (keeps softmax
// reductions off the LDS pipe; lane halves never mix, matching C layout).
__device__ __forceinline__ float permx(float x, int sel0, int sel1, int mask) {
#if __has_builtin(__builtin_amdgcn_permlane16)
    union { float f; u32 i; } c; c.f = x;
    c.i = __builtin_amdgcn_permlane16(c.i, c.i, sel0, sel1, false, false);
    return c.f;
#else
    return __shfl_xor(x, mask, 32);
#endif
}
#define XOR1(x) permx(x, (int)0x67452301, (int)0xEFCDAB89, 1)
#define XOR2(x) permx(x, (int)0x45670123, (int)0xCDEF89AB, 2)
#define XOR4(x) permx(x, (int)0x32107654, (int)0xBA98FEDC, 4)
#define XOR8(x) permx(x, (int)0xFEDCBA98, (int)0x76543210, 8)

__device__ __forceinline__ v8f wmma_bf16(v16bf a, v16bf b, v8f c) {
    return __builtin_amdgcn_wmma_f32_16x16x32_bf16(
        /*neg_a=*/false, a, /*neg_b=*/false, b,
        /*c_mod=*/(short)0, c, /*reuse_a=*/false, /*reuse_b=*/false);
}

// =====================================================================
// Kernel 1: qkv = tokens @ w_qkv.T ; split + RMSNorm ; emit bf16 operand
// tensors. Q is pre-scaled by DH^-0.5. V2 stored d-major (transposed).
// =====================================================================
__global__ void __launch_bounds__(256) proj_kernel(
    const float* __restrict__ tokens, const float* __restrict__ w_qkv,
    const float* __restrict__ qw, const float* __restrict__ k1w,
    const float* __restrict__ k2w, const float* __restrict__ v1w,
    const float* __restrict__ v2w,
    u16* __restrict__ Qbf, u16* __restrict__ K1bf, u16* __restrict__ K2bf,
    u16* __restrict__ V1bf, u16* __restrict__ V2t)
{
    __shared__ float tok[DIMSZ];
    __shared__ float row[QKV_COLS];
    __shared__ float rfac[40];

    const int bn = blockIdx.x;            // b*NSEQ + n
    const int b  = bn / NSEQ;
    const int n  = bn % NSEQ;
    const int tid = threadIdx.x;

    const float* tp = tokens + (size_t)bn * DIMSZ;
    for (int i = tid; i < DIMSZ; i += 256) tok[i] = tp[i];
    __syncthreads();

    for (int c = tid; c < QKV_COLS; c += 256) {
        const float* wr = w_qkv + (size_t)c * DIMSZ;
        float acc = 0.f;
        for (int d = 0; d < DIMSZ; d += 4) {
            float4 w4 = *(const float4*)(wr + d);
            acc += tok[d] * w4.x + tok[d + 1] * w4.y +
                   tok[d + 2] * w4.z + tok[d + 3] * w4.w;
        }
        row[c] = acc;
    }
    __syncthreads();

    if (tid < 40) {
        int base;
        if (tid < 8)       base = tid * 64;                                  // q heads
        else if (tid < 24) { int u = tid - 8;  base = 512  + (u >> 1) * 128 + (u & 1) * 64; }
        else               { int u = tid - 24; base = 1536 + (u >> 1) * 128 + (u & 1) * 64; }
        float ss = 0.f;
        for (int d = 0; d < DH; ++d) { float x = row[base + d]; ss += x * x; }
        rfac[tid] = rsqrtf(ss * (1.0f / DH) + 1.1920929e-07f);
    }
    __syncthreads();

    for (int c = tid; c < QKV_COLS; c += 256) {
        float val = row[c];
        if (c < 512) {
            int h = c >> 6, d = c & 63;
            float o = val * rfac[h] * qw[d] * 0.125f;   // fold logit scale
            Qbf[(((size_t)b * HEADS + h) * NSEQ + n) * DH + d] = f2bf(o);
        } else if (c < 1536) {
            int cc = c - 512, h = cc >> 7, wn = cc & 127;
            if (wn < 64)
                K1bf[(((size_t)b * HEADS + h) * NSEQ + n) * DH + wn] =
                    f2bf(val * rfac[8 + h * 2] * k1w[wn]);
            else {
                int d = wn - 64;
                K2bf[(((size_t)b * HEADS + h) * NSEQ + n) * DH + d] =
                    f2bf(val * rfac[8 + h * 2 + 1] * k2w[d]);
            }
        } else {
            int cc = c - 1536, h = cc >> 7, wn = cc & 127;
            if (wn < 64)
                V1bf[(((size_t)b * HEADS + h) * NSEQ + n) * DH + wn] =
                    f2bf(val * rfac[24 + h * 2] * v1w[wn]);
            else {
                int d = wn - 64;
                V2t[(((size_t)b * HEADS + h) * DH + d) * NSEQ + n] =  // transposed
                    f2bf(val * rfac[24 + h * 2 + 1] * v2w[d]);
            }
        }
    }
}

// =====================================================================
// Kernel 2: flash two-simplicial attention. One block = (b,h, 16-query
// tile). 4 waves split the j loop; per (j, 32 k's): synth keys/vals in
// LDS (v_pk_mul_bf16), 4 WMMAs logits + online softmax (v_permlane16
// butterflies) + 4 WMMAs PV. Wave partials merged via ds_add_f32.
// =====================================================================
__global__ void __launch_bounds__(128) attn_kernel(
    const u16* __restrict__ Qbf, const u16* __restrict__ K1bf,
    const u16* __restrict__ K2bf, const u16* __restrict__ V1bf,
    const u16* __restrict__ V2t, float* __restrict__ Attn)
{
    __shared__ u16  sKey[WAVES][KT][DH];    // [pair][d]  row-major
    __shared__ u16  sVT [WAVES][DH][KT];    // [d][pair]  d-major
    __shared__ u16  sP  [WAVES][16][KT];    // staged probabilities (bf16)
    __shared__ float sO[16][DH];
    __shared__ float sM[WAVES][16];
    __shared__ float sL[WAVES][16];
    __shared__ float sLtot[16];

    const int blk   = blockIdx.x;
    const int itile = blk % ITILES;
    const int bh    = blk / ITILES;
    const int tid   = threadIdx.x;
    const int wave  = tid >> 5;
    const int lane  = tid & 31;
    const int hf    = lane >> 4;    // lane half (0/1)
    const int ln    = lane & 15;

    const u16* Qp  = Qbf  + (size_t)bh * NSEQ * DH;
    const u16* K1p = K1bf + (size_t)bh * NSEQ * DH;
    const u16* K2p = K2bf + (size_t)bh * NSEQ * DH;
    const u16* V1p = V1bf + (size_t)bh * NSEQ * DH;
    const u16* V2p = V2t  + (size_t)bh * DH * NSEQ;

    // prefetch this (b,h)'s operand working set (~96KB) into cache
    {
        const size_t bytes = (size_t)NSEQ * DH * 2;     // 24576 per tensor
        for (size_t off = (size_t)tid * 128; off < bytes; off += 128 * 128) {
            __builtin_prefetch((const char*)K1p + off, 0, 3);
            __builtin_prefetch((const char*)K2p + off, 0, 3);
            __builtin_prefetch((const char*)V1p + off, 0, 3);
            __builtin_prefetch((const char*)V2p + off, 0, 3);
        }
    }

    for (int idx = tid; idx < 16 * DH; idx += 128) ((float*)sO)[idx] = 0.f;

    // ---- Q A-operands in registers (K-chunks d:[0,32),[32,64)) ----
    const int qrow = itile * 16 + ln;
    BfOp Qa[2];
#pragma unroll
    for (int ch = 0; ch < 2; ++ch) {
        const u16* base = Qp + qrow * DH + ch * 32 + hf * 16;
        Qa[ch].q[0] = *(const uint4*)(base);        // K = ch*32 + hf*16 + [0,8)
        Qa[ch].q[1] = *(const uint4*)(base + 8);    // K = ch*32 + hf*16 + [8,16)
    }

    float m[8], l[8];
    v8f O[4];
#pragma unroll
    for (int r = 0; r < 8; ++r) { m[r] = -3.0e38f; l[r] = 0.f; }
    v8f zero8 = {0.f, 0.f, 0.f, 0.f, 0.f, 0.f, 0.f, 0.f};
#pragma unroll
    for (int t = 0; t < 4; ++t) O[t] = zero8;

    for (int j = wave; j < NSEQ; j += WAVES) {
        // broadcast v1[j][d] as packed bf16x2 {v,v} for d=lane, lane+32
        u32 v1a = (u32)V1p[j * DH + lane];      v1a |= v1a << 16;
        u32 v1b = (u32)V1p[j * DH + lane + 32]; v1b |= v1b << 16;

        for (int k0 = 0; k0 < NSEQ; k0 += KT) {
            // ---- form Key tile: lane builds row k = k0+lane ----
            {
                const u16* k1r = K1p + j * DH;
                const u16* k2r = K2p + (k0 + lane) * DH;
                u16* dst = &sKey[wave][lane][0];
#pragma unroll
                for (int d = 0; d < DH; d += 8) {
                    uint4 a = *(const uint4*)(k1r + d);
                    uint4 b = *(const uint4*)(k2r + d);
                    *(uint4*)(dst + d) = pkmul8(a, b);
                }
            }
            // ---- form VT rows d = lane, lane+32 ----
#pragma unroll
            for (int rr = 0; rr < 2; ++rr) {
                const int d = lane + rr * 32;
                const u32 vs = rr ? v1b : v1a;
                const u16* v2r = V2p + d * NSEQ + k0;
                u16* dst = &sVT[wave][d][0];
#pragma unroll
                for (int k = 0; k < KT; k += 8) {
                    uint4 b = *(const uint4*)(v2r + k);
                    uint4 o;
                    o.x = pkmul_bf16(vs, b.x); o.y = pkmul_bf16(vs, b.y);
                    o.z = pkmul_bf16(vs, b.z); o.w = pkmul_bf16(vs, b.w);
                    *(uint4*)(dst + k) = o;
                }
            }
            // same-wave LDS store->load: DS ops are in-order per wave.

            // ---- logits S[16 x 32] = Q @ Key^T ----
            v8f S0 = zero8, S1 = zero8;
#pragma unroll
            for (int ch = 0; ch < 2; ++ch) {
                const int koff = ch * 32 + hf * 16;       // B: contiguous 16 K per half
                BfOp B0, B1;
                const u16* r0 = &sKey[wave][ln][koff];
                B0.q[0] = *(const uint4*)(r0);
                B0.q[1] = *(const uint4*)(r0 + 8);
                const u16* r1 = &sKey[wave][16 + ln][koff];
                B1.q[0] = *(const uint4*)(r1);
                B1.q[1] = *(const uint4*)(r1 + 8);
                S0 = wmma_bf16(Qa[ch].v, B0.v, S0);
                S1 = wmma_bf16(Qa[ch].v, B1.v, S1);
            }

            // ---- online softmax (rows r+8*hf live in VGPR r of this half) ----
#pragma unroll
            for (int r = 0; r < 8; ++r) {
                float tmax = fmaxf(S0[r], S1[r]);
                tmax = fmaxf(tmax, XOR1(tmax));
                tmax = fmaxf(tmax, XOR2(tmax));
                tmax = fmaxf(tmax, XOR4(tmax));
                tmax = fmaxf(tmax, XOR8(tmax));
                const float mn = fmaxf(m[r], tmax);
                const float alpha = __expf(m[r] - mn);
                const float p0 = __expf(S0[r] - mn);
                const float p1 = __expf(S1[r] - mn);
                float s = p0 + p1;
                s += XOR1(s);
                s += XOR2(s);
                s += XOR4(s);
                s += XOR8(s);
                l[r] = l[r] * alpha + s;
                m[r] = mn;
                O[0][r] *= alpha; O[1][r] *= alpha;
                O[2][r] *= alpha; O[3][r] *= alpha;
                const int rw = r + hf * 8;
                const u32 pk = cvt_pk_bf16(p0, p1);
                sP[wave][rw][ln]      = (u16)pk;
                sP[wave][rw][16 + ln] = (u16)(pk >> 16);
            }

            // ---- O[16 x 64] += P[16 x 32] @ V[32 x 64] ----
            BfOp Pa;
            const u16* pr = &sP[wave][ln][hf * 16];
            Pa.q[0] = *(const uint4*)(pr);
            Pa.q[1] = *(const uint4*)(pr + 8);
#pragma unroll
            for (int t = 0; t < 4; ++t) {
                BfOp Vb;
                const u16* vr = &sVT[wave][t * 16 + ln][hf * 16];
                Vb.q[0] = *(const uint4*)(vr);
                Vb.q[1] = *(const uint4*)(vr + 8);
                O[t] = wmma_bf16(Pa.v, Vb.v, O[t]);
            }
        }
    }

    // ---- merge 4 wave partials ----
    if (ln == 0) {   // lanes 0 (rows 0-7) and 16 (rows 8-15)
#pragma unroll
        for (int r = 0; r < 8; ++r) {
            sM[wave][r + hf * 8] = m[r];
            sL[wave][r + hf * 8] = l[r];
        }
    }
    __syncthreads();

    float sc[8];
#pragma unroll
    for (int r = 0; r < 8; ++r) {
        const int rw = r + hf * 8;
        float M = -3.0e38f;
        for (int w = 0; w < WAVES; ++w) M = fmaxf(M, sM[w][rw]);
        float lt = 0.f;
        for (int w = 0; w < WAVES; ++w) lt += sL[w][rw] * __expf(sM[w][rw] - M);
        sc[r] = __expf(m[r] - M);
        if (wave == 0 && ln == 0) sLtot[rw] = lt;
    }
#pragma unroll
    for (int t = 0; t < 4; ++t)
#pragma unroll
        for (int r = 0; r < 8; ++r)
            atomicAdd(&sO[r + hf * 8][t * 16 + ln], O[t][r] * sc[r]);
    __syncthreads();

    float* out = Attn + (size_t)bh * NSEQ * DH + (size_t)itile * 16 * DH;
    for (int idx = tid; idx < 16 * DH; idx += 128) {
        const int row = idx >> 6;
        out[idx] = sO[row][idx & 63] / sLtot[row];
    }
}

// =====================================================================
// Kernel 3: out = attn_flat @ w_out.T (fp32, tiny)
// =====================================================================
__global__ void __launch_bounds__(256) outproj_kernel(
    const float* __restrict__ Attn, const float* __restrict__ w_out,
    float* __restrict__ out)
{
    __shared__ float arow[DIMSZ];
    const int bn = blockIdx.x;
    const int b = bn / NSEQ, n = bn % NSEQ;
    const int tid = threadIdx.x;

    for (int c = tid; c < DIMSZ; c += 256) {
        const int h = c >> 6, d = c & 63;
        arow[c] = Attn[(((size_t)b * HEADS + h) * NSEQ + n) * DH + d];
    }
    __syncthreads();

    float* op = out + (size_t)bn * DIMSZ;
    for (int c = tid; c < DIMSZ; c += 256) {
        const float* wr = w_out + (size_t)c * DIMSZ;
        float acc = 0.f;
        for (int d = 0; d < DIMSZ; d += 4) {
            float4 w4 = *(const float4*)(wr + d);
            acc += arow[d] * w4.x + arow[d + 1] * w4.y +
                   arow[d + 2] * w4.z + arow[d + 3] * w4.w;
        }
        op[c] = acc;
    }
}

extern "C" void kernel_launch(void* const* d_in, const int* in_sizes, int n_in,
                              void* d_out, int out_size, void* d_ws, size_t ws_size,
                              hipStream_t stream) {
    const float* tokens = (const float*)d_in[0];
    const float* w_qkv  = (const float*)d_in[1];
    const float* w_out  = (const float*)d_in[2];
    const float* qw     = (const float*)d_in[3];
    const float* k1w    = (const float*)d_in[4];
    const float* k2w    = (const float*)d_in[5];
    const float* v1w    = (const float*)d_in[6];
    const float* v2w    = (const float*)d_in[7];

    char* ws = (char*)d_ws;
    const size_t TEN = (size_t)BATCH * HEADS * NSEQ * DH;   // 196608 elems
    u16* Qbf   = (u16*)(ws);
    u16* K1bf  = (u16*)(ws + TEN * 2);
    u16* K2bf  = (u16*)(ws + TEN * 4);
    u16* V1bf  = (u16*)(ws + TEN * 6);
    u16* V2t   = (u16*)(ws + TEN * 8);
    float* Attn = (float*)(ws + TEN * 10);   // total ws use: TEN*14 ≈ 2.75 MB

    proj_kernel<<<BATCH * NSEQ, 256, 0, stream>>>(
        tokens, w_qkv, qw, k1w, k2w, v1w, v2w, Qbf, K1bf, K2bf, V1bf, V2t);
    attn_kernel<<<BATCH * HEADS * ITILES, 128, 0, stream>>>(
        Qbf, K1bf, K2bf, V1bf, V2t, Attn);
    outproj_kernel<<<BATCH * NSEQ, 256, 0, stream>>>(Attn, w_out, (float*)d_out);
}
